// InferotemporalCortex_84009560310048
// MI455X (gfx1250) — matrix operand order repaired and checked by reading, not compiled, not run on hardware
//
#include <hip/hip_runtime.h>
#include <hip/hip_bf16.h>

// ---------------------------------------------------------------------------
// CDNA5 (gfx1250) implementation.
// Internal activation format: NHWC bf16 so WMMA fragments load with plain
// b128 global loads (K contiguous per lane). GEMMs use
// V_WMMA_F32_16X16X32_BF16 with f32 accumulation; BN+ReLU fused in epilogue.
// ---------------------------------------------------------------------------

typedef __attribute__((ext_vector_type(16))) __bf16 bf16x16;
typedef __attribute__((ext_vector_type(8)))  __bf16 bf16x8;
typedef __attribute__((ext_vector_type(8)))  float  f32x8;

union Frag16 { bf16x16 v; bf16x8 h[2]; };

__device__ __forceinline__ unsigned short f2bf(float f) {
  unsigned int u = __float_as_uint(f);
  u += 0x7FFFu + ((u >> 16) & 1u);           // round-to-nearest-even
  return (unsigned short)(u >> 16);
}
__device__ __forceinline__ float bf2f(unsigned short h) {
  return __uint_as_float(((unsigned int)h) << 16);
}
__device__ __forceinline__ bf16x8 ld8(const unsigned short* p) {
  return *reinterpret_cast<const bf16x8*>(p);
}

// ---------------- small prep kernels ----------------------------------------
__global__ void cvt_bf16(const float* __restrict__ src,
                         unsigned short* __restrict__ dst, int n) {
  int i = blockIdx.x * 256 + threadIdx.x;
  if (i < n) dst[i] = f2bf(src[i]);
}

__global__ void bn_prep(const float* __restrict__ g, const float* __restrict__ b,
                        const float* __restrict__ m, const float* __restrict__ v,
                        float* __restrict__ s, float* __restrict__ t, int n) {
  int i = blockIdx.x * 256 + threadIdx.x;
  if (i < n) {
    float sc = g[i] * rsqrtf(v[i] + 1e-5f);
    s[i] = sc;
    t[i] = b[i] - m[i] * sc;
  }
}

// ---------------- depthwise 3x3 #1: NCHW f32 -> NHWC bf16 -------------------
// x: [32][128][64][64] f32, w: [128][9], out: [32][4096][128] bf16
__global__ __launch_bounds__(256) void dw1_kernel(
    const float* __restrict__ x, const float* __restrict__ w,
    unsigned short* __restrict__ out) {
  int idx = blockIdx.x * 256 + threadIdx.x;      // 32*4096*128 threads
  int c = idx & 127;
  int p = (idx >> 7) & 4095;
  int b = idx >> 19;
  int h = p >> 6, wq = p & 63;
  const float* xb = x + ((size_t)b * 128 + c) * 4096;
  const float* wc = w + c * 9;
  float acc = 0.f;
#pragma unroll
  for (int dh = -1; dh <= 1; ++dh) {
    int hh = h + dh;
    if (hh < 0 || hh > 63) continue;
#pragma unroll
    for (int dw = -1; dw <= 1; ++dw) {
      int ww = wq + dw;
      if (ww < 0 || ww > 63) continue;
      acc += xb[hh * 64 + ww] * wc[(dh + 1) * 3 + (dw + 1)];
    }
  }
  out[idx] = f2bf(acc);
}

// ---------------- depthwise 3x3 #2: NHWC bf16 -> NHWC bf16 ------------------
// in: [32][1024][256] (32x32 spatial), w: [256][9], out same shape
__global__ __launch_bounds__(256) void dw2_kernel(
    const unsigned short* __restrict__ in, const float* __restrict__ w,
    unsigned short* __restrict__ out) {
  int idx = blockIdx.x * 256 + threadIdx.x;      // 32*1024*256 threads
  int c = idx & 255;
  int p = (idx >> 8) & 1023;
  int b = idx >> 18;
  int h = p >> 5, wq = p & 31;
  const unsigned short* ib = in + (size_t)b * 1024 * 256;
  const float* wc = w + c * 9;
  float acc = 0.f;
#pragma unroll
  for (int dh = -1; dh <= 1; ++dh) {
    int hh = h + dh;
    if (hh < 0 || hh > 31) continue;
#pragma unroll
    for (int dw = -1; dw <= 1; ++dw) {
      int ww = wq + dw;
      if (ww < 0 || ww > 31) continue;
      acc += bf2f(ib[(size_t)(hh * 32 + ww) * 256 + c]) * wc[(dh + 1) * 3 + (dw + 1)];
    }
  }
  out[idx] = f2bf(acc);
}

// ---------------- maxpool 2x2 (NHWC bf16), 64x64 -> 32x32 -------------------
__global__ __launch_bounds__(256) void pool1_kernel(
    const unsigned short* __restrict__ in, unsigned short* __restrict__ out) {
  int idx = blockIdx.x * 256 + threadIdx.x;      // 32*1024*256 threads
  int c = idx & 255;
  int p = (idx >> 8) & 1023;
  int b = idx >> 18;
  int ho = p >> 5, wo = p & 31;
  const unsigned short* base =
      in + ((size_t)b * 4096 + (size_t)(2 * ho) * 64 + 2 * wo) * 256 + c;
  float m0 = bf2f(base[0]);
  float m1 = bf2f(base[256]);
  float m2 = bf2f(base[64 * 256]);
  float m3 = bf2f(base[64 * 256 + 256]);
  out[idx] = f2bf(fmaxf(fmaxf(m0, m1), fmaxf(m2, m3)));
}

// ---------------- maxpool 2x2: 32x32 -> 16x16; dual output ------------------
// xF: f32 NCHW [32][256][256] (d_out "x"), xB: bf16 NHWC [32][256][256]
__global__ __launch_bounds__(256) void pool2_kernel(
    const unsigned short* __restrict__ in, float* __restrict__ xF,
    unsigned short* __restrict__ xB) {
  int idx = blockIdx.x * 256 + threadIdx.x;      // 32*256*256 threads
  int c = idx & 255;
  int p = (idx >> 8) & 255;
  int b = idx >> 16;
  int ho = p >> 4, wo = p & 15;
  const unsigned short* base =
      in + ((size_t)b * 1024 + (size_t)(2 * ho) * 32 + 2 * wo) * 256 + c;
  float m0 = bf2f(base[0]);
  float m1 = bf2f(base[256]);
  float m2 = bf2f(base[32 * 256]);
  float m3 = bf2f(base[32 * 256 + 256]);
  float m = fmaxf(fmaxf(m0, m1), fmaxf(m2, m3));
  xB[idx] = f2bf(m);
  xF[((size_t)b * 256 + c) * 256 + p] = m;
}

// ---------------- WMMA GEMM -------------------------------------------------
// A: [B][M][K] bf16 (NHWC activations), Bw: [N][K] bf16 (weights)
// MODE 0: out = relu(scale[n]*acc + bias[n]) -> bf16 NHWC [B][M][N]
// MODE 1: out = acc -> f32 "NCHW" [B][N][M]
// Block: 256 threads = 8 waves; wave tile 32x32 (2x2 WMMA); block tile 128x64.
template <int MODE, int KDIM>
__global__ __launch_bounds__(256) void gemm_wmma(
    const unsigned short* __restrict__ A, const unsigned short* __restrict__ Bw,
    float* __restrict__ outF, unsigned short* __restrict__ outB,
    const float* __restrict__ scale, const float* __restrict__ bias, int M,
    int N) {
  const int b    = blockIdx.z;
  const int lane = threadIdx.x & 31;
  const int wave = threadIdx.x >> 5;
  const int wm   = wave & 3;
  const int wn   = wave >> 2;
  const int m0   = blockIdx.x * 128 + wm * 32;
  const int n0   = blockIdx.y * 64 + wn * 32;
  const int lm   = lane & 15;
  const int hi   = lane >> 4;

  const unsigned short* Ab = A + (size_t)b * M * KDIM;

  f32x8 acc[2][2] = {};
  const int arow = m0 + lm;      // + mi*16
  const int bcol = n0 + lm;      // + ni*16
  const int kA   = hi * 8;       // A lane K base offset within 32-step
  const int kB   = hi * 8;       // B lane K base offset within 32-step

#pragma unroll
  for (int k0 = 0; k0 < KDIM; k0 += 32) {
    Frag16 af[2], bf[2];
#pragma unroll
    for (int mi = 0; mi < 2; ++mi) {
      const unsigned short* p = Ab + (size_t)(arow + mi * 16) * KDIM + k0 + kA;
      af[mi].h[0] = ld8(p);        // K = kbase + 0..7
      af[mi].h[1] = ld8(p + 16);   // K = kbase + 16..23
    }
#pragma unroll
    for (int ni = 0; ni < 2; ++ni) {
      const unsigned short* p = Bw + (size_t)(bcol + ni * 16) * KDIM + k0 + kB;
      bf[ni].h[0] = ld8(p);        // K = kb + 0..7
      bf[ni].h[1] = ld8(p + 16);   // K = kb + 16..23
    }
#pragma unroll
    for (int mi = 0; mi < 2; ++mi)
#pragma unroll
      for (int ni = 0; ni < 2; ++ni)
        acc[mi][ni] = __builtin_amdgcn_wmma_f32_16x16x32_bf16(
            false, af[mi].v, false, bf[ni].v, (short)0, acc[mi][ni], false,
            false);
  }

  if (MODE == 0) {
    unsigned short* Ob = outB + (size_t)b * M * N;
#pragma unroll
    for (int ni = 0; ni < 2; ++ni) {
      int n = n0 + ni * 16 + lm;
      float s = scale[n];
      float t = bias[n];
#pragma unroll
      for (int mi = 0; mi < 2; ++mi) {
        int mbase = m0 + mi * 16 + hi * 8;
#pragma unroll
        for (int r = 0; r < 8; ++r) {
          float v = fmaxf(acc[mi][ni][r] * s + t, 0.0f);
          Ob[(size_t)(mbase + r) * N + n] = f2bf(v);
        }
      }
    }
  } else {
    float* Of = outF + (size_t)b * N * M;
#pragma unroll
    for (int ni = 0; ni < 2; ++ni) {
      int n = n0 + ni * 16 + lm;
#pragma unroll
      for (int mi = 0; mi < 2; ++mi) {
        int mbase = m0 + mi * 16 + hi * 8;
        float4 v0 = make_float4(acc[mi][ni][0], acc[mi][ni][1],
                                acc[mi][ni][2], acc[mi][ni][3]);
        float4 v1 = make_float4(acc[mi][ni][4], acc[mi][ni][5],
                                acc[mi][ni][6], acc[mi][ni][7]);
        *reinterpret_cast<float4*>(Of + (size_t)n * M + mbase)     = v0;
        *reinterpret_cast<float4*>(Of + (size_t)n * M + mbase + 4) = v1;
      }
    }
  }
}

// ---------------- final reductions ------------------------------------------
// x:   f32 NCHW [32][256][256]   (d_out "x" region)
// cat: f32      [32][256][256]   (d_out cat_spikes region, channel = k*8+o)
__global__ __launch_bounds__(256) void finalize_kernel(
    const float* __restrict__ x, const float* __restrict__ cat,
    const float* __restrict__ emb_w, const float* __restrict__ emb_b,
    float* __restrict__ catAct, float* __restrict__ obj) {
  __shared__ float pooled[256];
  int b = blockIdx.x, t = threadIdx.x;
  const float* xb = x + ((size_t)b * 256 + t) * 256;
  float s = 0.f;
  for (int i = 0; i < 256; ++i) s += xb[i];
  pooled[t] = s * (1.f / 256.f);
  __syncthreads();
  if (t < 32) {
    const float* cb = cat + ((size_t)b * 256 + t * 8) * 256;
    float a = 0.f;
    for (int i = 0; i < 8 * 256; ++i) a += cb[i];
    catAct[b * 32 + t] = a * (1.f / 2048.f);
  }
  if (t < 128) {
    float a = emb_b[t];
    const float* wr = emb_w + (size_t)t * 256;
    for (int c = 0; c < 256; ++c) a += pooled[c] * wr[c];
    obj[b * 128 + t] = a;
  }
}

// ---------------------------------------------------------------------------
extern "C" void kernel_launch(void* const* d_in, const int* in_sizes, int n_in,
                              void* d_out, int out_size, void* d_ws,
                              size_t ws_size, hipStream_t stream) {
  (void)in_sizes; (void)n_in; (void)out_size; (void)ws_size;

  const float* v4    = (const float*)d_in[0];
  const float* w1_dw = (const float*)d_in[1];
  const float* w1_pw = (const float*)d_in[2];
  const float* bn1_g = (const float*)d_in[3];
  const float* bn1_b = (const float*)d_in[4];
  const float* bn1_m = (const float*)d_in[5];
  const float* bn1_v = (const float*)d_in[6];
  const float* w2_dw = (const float*)d_in[7];
  const float* w2_pw = (const float*)d_in[8];
  const float* bn2_g = (const float*)d_in[9];
  const float* bn2_b = (const float*)d_in[10];
  const float* bn2_m = (const float*)d_in[11];
  const float* bn2_v = (const float*)d_in[12];
  const float* cat_w = (const float*)d_in[13];
  const float* emb_w = (const float*)d_in[14];
  const float* emb_b = (const float*)d_in[15];

  // d_out layout (floats): cat_spikes | category_activations | embedding | x
  float* outCat = (float*)d_out;                 // 32*32*8*16*16 = 2097152
  float* outAct = outCat + 2097152;              // 1024
  float* outEmb = outAct + 1024;                 // 4096
  float* outX   = outEmb + 4096;                 // 2097152

  // workspace carve-up
  char* w = (char*)d_ws;
  unsigned short* a1  = (unsigned short*)w; w += (size_t)32 * 4096 * 128 * 2;
  unsigned short* y1  = (unsigned short*)w; w += (size_t)32 * 4096 * 256 * 2;
  unsigned short* p1  = (unsigned short*)w; w += (size_t)32 * 1024 * 256 * 2;
  unsigned short* a2  = (unsigned short*)w; w += (size_t)32 * 1024 * 256 * 2;
  unsigned short* y2  = (unsigned short*)w; w += (size_t)32 * 1024 * 256 * 2;
  unsigned short* x2b = (unsigned short*)w; w += (size_t)32 * 256 * 256 * 2;
  unsigned short* w1b = (unsigned short*)w; w += (size_t)256 * 128 * 2;
  unsigned short* w2b = (unsigned short*)w; w += (size_t)256 * 256 * 2;
  unsigned short* wcb = (unsigned short*)w; w += (size_t)256 * 256 * 2;
  float* s1 = (float*)w; w += 1024;
  float* t1 = (float*)w; w += 1024;
  float* s2 = (float*)w; w += 1024;
  float* t2 = (float*)w; w += 1024;

  // weight conversion + BN folding
  cvt_bf16<<<128, 256, 0, stream>>>(w1_pw, w1b, 256 * 128);
  cvt_bf16<<<256, 256, 0, stream>>>(w2_pw, w2b, 256 * 256);
  cvt_bf16<<<256, 256, 0, stream>>>(cat_w, wcb, 256 * 256);
  bn_prep<<<1, 256, 0, stream>>>(bn1_g, bn1_b, bn1_m, bn1_v, s1, t1, 256);
  bn_prep<<<1, 256, 0, stream>>>(bn2_g, bn2_b, bn2_m, bn2_v, s2, t2, 256);

  // block 1
  dw1_kernel<<<65536, 256, 0, stream>>>(v4, w1_dw, a1);
  gemm_wmma<0, 128><<<dim3(32, 4, 32), 256, 0, stream>>>(a1, w1b, nullptr, y1,
                                                         s1, t1, 4096, 256);
  pool1_kernel<<<32768, 256, 0, stream>>>(y1, p1);

  // block 2
  dw2_kernel<<<32768, 256, 0, stream>>>(p1, w2_dw, a2);
  gemm_wmma<0, 256><<<dim3(8, 4, 32), 256, 0, stream>>>(a2, w2b, nullptr, y2,
                                                        s2, t2, 1024, 256);
  pool2_kernel<<<8192, 256, 0, stream>>>(y2, outX, x2b);

  // cat_spikes (1x1 conv, 256->256), f32 NCHW straight into d_out
  gemm_wmma<1, 256><<<dim3(2, 4, 32), 256, 0, stream>>>(x2b, wcb, outCat,
                                                        nullptr, s2, t2, 256,
                                                        256);

  // pooled means + embedding GEMV
  finalize_kernel<<<32, 256, 0, stream>>>(outX, outCat, emb_w, emb_b, outAct,
                                          outEmb);
}
